// Position_attention_38689065402392
// MI455X (gfx1250) — matrix operand order repaired and checked by reading, not compiled, not run on hardware
//
#include <hip/hip_runtime.h>
#include <hip/hip_bf16.h>

#define N_POS 8000
#define C_DIM 128
#define WT_STRIDE 136      // padded LDS row stride for weight tiles (272B)
#define BK 32              // keys staged per iteration
#define KT_STRIDE 136      // bf16 elems per K-tile LDS row: 256B data + 16B pad
#define VT_LDS_STRIDE 40   // bf16 elems per V-tile LDS row: 64B data + 16B pad

typedef __attribute__((ext_vector_type(16))) __bf16 v16bf;
typedef __attribute__((ext_vector_type(8)))  __bf16 v8bf;
typedef __attribute__((ext_vector_type(8)))  float  v8f;
typedef __attribute__((ext_vector_type(4)))  float  f32x4;

#if defined(__has_builtin)
#if __has_builtin(__builtin_amdgcn_tensor_load_to_lds) && \
    __has_builtin(__builtin_amdgcn_s_wait_tensorcnt)
#define HAVE_TDM 1
#endif
#endif
#ifndef HAVE_TDM
#define HAVE_TDM 0
#endif

static __device__ __forceinline__ v16bf join16(v8bf lo, v8bf hi) {
  return __builtin_shufflevector(lo, hi, 0,1,2,3,4,5,6,7,8,9,10,11,12,13,14,15);
}

static __device__ __forceinline__ v8f wmma_bf16(v16bf a, v16bf b, v8f c) {
  return __builtin_amdgcn_wmma_f32_16x16x32_bf16(false, a, false, b, (short)0, c,
                                                 false, false);
}

static __device__ __forceinline__ v8bf cvt8(const float* __restrict__ p) {
  f32x4 a = *(const f32x4*)p;
  f32x4 b = *(const f32x4*)(p + 4);
  v8bf r;
#pragma unroll
  for (int i = 0; i < 4; ++i) { r[i] = (__bf16)a[i]; r[i + 4] = (__bf16)b[i]; }
  return r;
}

static __device__ __forceinline__ v8bf mulcvt8(const float* __restrict__ pa,
                                               const float* __restrict__ pb) {
  f32x4 a0 = *(const f32x4*)pa, a1 = *(const f32x4*)(pa + 4);
  f32x4 b0 = *(const f32x4*)pb, b1 = *(const f32x4*)(pb + 4);
  v8bf r;
#pragma unroll
  for (int i = 0; i < 4; ++i) {
    r[i]     = (__bf16)(a0[i] * b0[i]);
    r[i + 4] = (__bf16)(a1[i] * b1[i]);
  }
  return r;
}

#if HAVE_TDM
typedef __attribute__((ext_vector_type(4))) unsigned int u32x4;
typedef __attribute__((ext_vector_type(8))) int          i32x8;
typedef __attribute__((ext_vector_type(4))) int          i32x4;

// Issue one TDM descriptor: 2D tile (tile0 x tile1) of 8-byte elements with
// LDS row padding (pad after every (1<<pi) dwords... per ISA: after 2<<pi
// dwords, add (pa+1) dwords).  1D loads use tile1 == 1.
static __device__ __forceinline__ void tdm_load(
    unsigned int lds_addr, const void* gaddr, unsigned int tile0,
    unsigned int tile1, unsigned long long stride0_e8, unsigned int pi,
    unsigned int pa) {
  const unsigned long long ga = (unsigned long long)(size_t)gaddr;
  u32x4 g0;
  g0[0] = 1u;                                        // count=1, user desc
  g0[1] = lds_addr;                                  // lds_addr [63:32]
  g0[2] = (unsigned int)ga;                          // global_addr lo
  g0[3] = (unsigned int)((ga >> 32) & 0x1FFFFFFu) | (2u << 30);  // hi + type=2

  const unsigned int td0 = (unsigned int)stride0_e8; // tensor_dim0 >= tile0
  const unsigned int td1 = tile1;
  i32x8 g1;
  g1[0] = (int)((3u << 16) | (1u << 20) | (pi << 22) | (pa << 25));
  g1[1] = (int)((td0 & 0xFFFFu) << 16);                       // td0[15:0] @63:48
  g1[2] = (int)(((td0 >> 16) & 0xFFFFu) | ((td1 & 0xFFFFu) << 16));
  g1[3] = (int)(((td1 >> 16) & 0xFFFFu) | ((tile0 & 0xFFFFu) << 16));
  g1[4] = (int)(tile1 & 0xFFFFu);                             // tile_dim1; dim2=0
  g1[5] = (int)(unsigned int)stride0_e8;                      // stride0 lo
  g1[6] = (int)(unsigned int)((stride0_e8 >> 32) & 0xFFFFu);  // stride0 hi
  g1[7] = 0;
  i32x4 z4 = {0, 0, 0, 0};
#if __clang_major__ >= 23
  i32x8 z8 = {0, 0, 0, 0, 0, 0, 0, 0};
  __builtin_amdgcn_tensor_load_to_lds(g0, g1, z4, z4, z8, 0);
#else
  __builtin_amdgcn_tensor_load_to_lds(g0, g1, z4, z4, 0);
#endif
}
#endif  // HAVE_TDM

// ---------------------------------------------------------------------------
// Kernel 1: pointwise projection  out = x @ W + b
// ---------------------------------------------------------------------------
__global__ __launch_bounds__(256) void proj_kernel(
    const float* __restrict__ x, const float* __restrict__ W,
    const float* __restrict__ b, __bf16* __restrict__ out, int transposeOut) {
  __shared__ __bf16 wt[C_DIM * WT_STRIDE];
  const int tid = threadIdx.x;
  for (int i = tid; i < C_DIM * C_DIM; i += 256) {
    int k = i >> 7, c = i & (C_DIM - 1);
    wt[c * WT_STRIDE + k] = (__bf16)W[i];
  }
  __syncthreads();

  const int wave = tid >> 5, lane = tid & 31;
  const int l16 = lane & 15;
  const bool hiHalf = lane >= 16;
  const int rowBase = blockIdx.x * 128 + wave * 16;
  if (rowBase >= N_POS) return;

  v16bf A[4];
  {
    const float* xr = x + (size_t)(rowBase + l16) * C_DIM;
#pragma unroll
    for (int kc = 0; kc < 4; ++kc) {
      int base = kc * 32 + (hiHalf ? 8 : 0);
      A[kc] = join16(cvt8(xr + base), cvt8(xr + base + 16));
    }
  }

#pragma unroll
  for (int cb = 0; cb < 8; ++cb) {
    const int chan = cb * 16 + l16;
    v8f acc;
    const float bias = b[chan];
#pragma unroll
    for (int r = 0; r < 8; ++r) acc[r] = bias;
#pragma unroll
    for (int kc = 0; kc < 4; ++kc) {
      const __bf16* wp = &wt[chan * WT_STRIDE + kc * 32 + (hiHalf ? 8 : 0)];
      v16bf B = join16(*(const v8bf*)wp, *(const v8bf*)(wp + 16));
      acc = wmma_bf16(A[kc], B, acc);
    }
    if (transposeOut) {
      v8bf pk;
#pragma unroll
      for (int r = 0; r < 8; ++r) pk[r] = (__bf16)acc[r];
      *(v8bf*)&out[(size_t)chan * N_POS + rowBase + (hiHalf ? 8 : 0)] = pk;
    } else {
#pragma unroll
      for (int r = 0; r < 8; ++r) {
        int row = rowBase + r + (hiHalf ? 8 : 0);
        out[(size_t)row * C_DIM + chan] = (__bf16)acc[r];
      }
    }
  }
}

// ---------------------------------------------------------------------------
// Kernel 2: flash attention with TDM-staged K / V^T tiles (double buffered)
// ---------------------------------------------------------------------------
__global__ __launch_bounds__(256) void attn_kernel(
    const __bf16* __restrict__ q, const __bf16* __restrict__ vT,
    float* __restrict__ ctx) {
  __shared__ __bf16 kt[2][BK * KT_STRIDE];          // 2 x 8704 B
  __shared__ __bf16 vt[2][C_DIM * VT_LDS_STRIDE];   // 2 x 10240 B

  const int tid = threadIdx.x, wave = tid >> 5, lane = tid & 31;
  const int l16 = lane & 15;
  const bool hiHalf = lane >= 16;
  const int qb = blockIdx.x * 128 + wave * 16;
  const bool active = qb < N_POS;
  const int qrow = active ? (qb + l16) : l16;

  // Q resident in B-layout for the whole key loop.
  v16bf Qb[4];
  {
    const __bf16* qr = q + (size_t)qrow * C_DIM;
#pragma unroll
    for (int kc = 0; kc < 4; ++kc) {
      const __bf16* p = qr + kc * 32 + (hiHalf ? 8 : 0);
      Qb[kc] = join16(*(const v8bf*)p, *(const v8bf*)(p + 16));
    }
  }

  v8f acc[8];
#pragma unroll
  for (int cb = 0; cb < 8; ++cb)
#pragma unroll
    for (int r = 0; r < 8; ++r) acc[cb][r] = 0.f;

  float m_i = -3.0e38f, l_i = 0.f;
  const int NITER = N_POS / BK;  // 250

#if HAVE_TDM
  // Prologue: stage tile 0 into buffer 0.
  if (wave == 0) {
    // K tile: BK rows x 256B, contiguous in q; 1D, pad 16B per 256B (272B rows)
    tdm_load((unsigned int)(size_t)&kt[0][0], q, (BK * C_DIM * 2) / 8, 1,
             (BK * C_DIM * 2) / 8, 5u, 3u);
    // V^T tile: 128 rows x 64B, row stride N*2 B; pad 16B per 64B (80B rows)
    tdm_load((unsigned int)(size_t)&vt[0][0], vT, (BK * 2) / 8, C_DIM,
             (unsigned long long)(N_POS * 2) / 8, 3u, 3u);
    __builtin_amdgcn_s_wait_tensorcnt(0);
  }
  __syncthreads();
#endif

  for (int it = 0; it < NITER; ++it) {
    const int cur = it & 1;

#if HAVE_TDM
    if (wave == 0 && it + 1 < NITER) {
      const int kb = (it + 1) * BK;
      tdm_load((unsigned int)(size_t)&kt[cur ^ 1][0],
               q + (size_t)kb * C_DIM, (BK * C_DIM * 2) / 8, 1,
               (BK * C_DIM * 2) / 8, 5u, 3u);
      tdm_load((unsigned int)(size_t)&vt[cur ^ 1][0], vT + kb, (BK * 2) / 8,
               C_DIM, (unsigned long long)(N_POS * 2) / 8, 3u, 3u);
    }
#else
    // Fallback: cooperative single-buffered copy with normal loads.
    {
      const int kb = it * BK;
      __syncthreads();
      {  // K: 32 rows x 256B; 8 threads/row x 32B
        int row = tid >> 3, col = (tid & 7) * 16;
        const __bf16* src = q + (size_t)(kb + row) * C_DIM + col;
        __bf16* dst = &kt[0][row * KT_STRIDE + col];
        *(v8bf*)dst = *(const v8bf*)src;
        *(v8bf*)(dst + 8) = *(const v8bf*)(src + 8);
      }
      {  // V^T: 128 rows x 64B; 2 threads/row x 32B
        int row = tid >> 1, col = (tid & 1) * 16;
        const __bf16* src = vT + (size_t)row * N_POS + kb + col;
        __bf16* dst = &vt[0][row * VT_LDS_STRIDE + col];
        *(v8bf*)dst = *(const v8bf*)src;
        *(v8bf*)(dst + 8) = *(const v8bf*)(src + 8);
      }
      __syncthreads();
    }
#endif
    const int cb_idx = HAVE_TDM ? cur : 0;
    const __bf16* ktb = &kt[cb_idx][0];
    const __bf16* vtb = &vt[cb_idx][0];

    // ---- S^T tiles from LDS (conflict-free: 272B row stride) ----
    v8f s0, s1;
#pragma unroll
    for (int r = 0; r < 8; ++r) { s0[r] = 0.f; s1[r] = 0.f; }
#pragma unroll
    for (int kc = 0; kc < 4; ++kc) {
      const int base = kc * 32 + (hiHalf ? 8 : 0);
      const __bf16* k0 = ktb + l16 * KT_STRIDE + base;
      v16bf A0 = join16(*(const v8bf*)k0, *(const v8bf*)(k0 + 16));
      s0 = wmma_bf16(A0, Qb[kc], s0);
      const __bf16* k1 = ktb + (16 + l16) * KT_STRIDE + base;
      v16bf A1 = join16(*(const v8bf*)k1, *(const v8bf*)(k1 + 16));
      s1 = wmma_bf16(A1, Qb[kc], s1);
    }

    // ---- online softmax (per-lane stats + one cross-half shuffle) ----
    float pm = s0[0];
#pragma unroll
    for (int r = 1; r < 8; ++r) pm = fmaxf(pm, s0[r]);
#pragma unroll
    for (int r = 0; r < 8; ++r) pm = fmaxf(pm, s1[r]);
    pm = fmaxf(pm, __shfl_xor(pm, 16, 32));
    const float mnew = fmaxf(m_i, pm);
    const float scale = __expf(m_i - mnew);

    float p0[8], p1[8], ps = 0.f;
#pragma unroll
    for (int r = 0; r < 8; ++r) {
      p0[r] = __expf(s0[r] - mnew);
      p1[r] = __expf(s1[r] - mnew);
      ps += p0[r] + p1[r];
    }
    ps += __shfl_xor(ps, 16, 32);
    l_i = l_i * scale + ps;
    m_i = mnew;

#pragma unroll
    for (int cb = 0; cb < 8; ++cb)
#pragma unroll
      for (int r = 0; r < 8; ++r) acc[cb][r] *= scale;

    // ---- P^T into B-layout ----
    v16bf Pb;
#pragma unroll
    for (int j = 0; j < 8; ++j) {
      float e0 = __shfl_xor(p0[j], 16, 32);
      float e1 = __shfl_xor(p1[j], 16, 32);
      Pb[j]     = (__bf16)(hiHalf ? e1 : p0[j]);
      Pb[j + 8] = (__bf16)(hiHalf ? p1[j] : e0);
    }

    // ---- ctx^T += V^T . P^T from LDS (conflict-free: 80B row stride) ----
#pragma unroll
    for (int cb = 0; cb < 8; ++cb) {
      const __bf16* vp = vtb + (cb * 16 + l16) * VT_LDS_STRIDE + (hiHalf ? 16 : 0);
      v16bf Av = join16(*(const v8bf*)vp, *(const v8bf*)(vp + 8));
      acc[cb] = wmma_bf16(Av, Pb, acc[cb]);
    }

#if HAVE_TDM
    if (wave == 0 && it + 1 < NITER) __builtin_amdgcn_s_wait_tensorcnt(0);
    __syncthreads();
#endif
  }

  if (active) {
    const float inv = 1.f / l_i;
    const int row = qb + l16;
#pragma unroll
    for (int cb = 0; cb < 8; ++cb)
#pragma unroll
      for (int r = 0; r < 8; ++r) {
        int chan = cb * 16 + r + (hiHalf ? 8 : 0);
        ctx[(size_t)row * C_DIM + chan] = acc[cb][r] * inv;
      }
  }
}

// ---------------------------------------------------------------------------
// Kernel 3: out = (x * ctx) @ W3 + b3
// ---------------------------------------------------------------------------
__global__ __launch_bounds__(256) void out_kernel(
    const float* __restrict__ x, const float* __restrict__ ctx,
    const float* __restrict__ W3, const float* __restrict__ b3,
    float* __restrict__ out) {
  __shared__ __bf16 wt[C_DIM * WT_STRIDE];
  const int tid = threadIdx.x;
  for (int i = tid; i < C_DIM * C_DIM; i += 256) {
    int k = i >> 7, c = i & (C_DIM - 1);
    wt[c * WT_STRIDE + k] = (__bf16)W3[i];
  }
  __syncthreads();

  const int wave = tid >> 5, lane = tid & 31;
  const int l16 = lane & 15;
  const bool hiHalf = lane >= 16;
  const int rowBase = blockIdx.x * 128 + wave * 16;
  if (rowBase >= N_POS) return;

  v16bf A[4];
  {
    const float* xr = x + (size_t)(rowBase + l16) * C_DIM;
    const float* cr = ctx + (size_t)(rowBase + l16) * C_DIM;
#pragma unroll
    for (int kc = 0; kc < 4; ++kc) {
      int base = kc * 32 + (hiHalf ? 8 : 0);
      A[kc] = join16(mulcvt8(xr + base, cr + base),
                     mulcvt8(xr + base + 16, cr + base + 16));
    }
  }

#pragma unroll
  for (int cb = 0; cb < 8; ++cb) {
    const int chan = cb * 16 + l16;
    v8f acc;
    const float bias = b3[chan];
#pragma unroll
    for (int r = 0; r < 8; ++r) acc[r] = bias;
#pragma unroll
    for (int kc = 0; kc < 4; ++kc) {
      const __bf16* wp = &wt[chan * WT_STRIDE + kc * 32 + (hiHalf ? 8 : 0)];
      v16bf B = join16(*(const v8bf*)wp, *(const v8bf*)(wp + 16));
      acc = wmma_bf16(A[kc], B, acc);
    }
#pragma unroll
    for (int r = 0; r < 8; ++r) {
      int row = rowBase + r + (hiHalf ? 8 : 0);
      out[(size_t)row * C_DIM + chan] = acc[r];
    }
  }
}

extern "C" void kernel_launch(void* const* d_in, const int* in_sizes, int n_in,
                              void* d_out, int out_size, void* d_ws,
                              size_t ws_size, hipStream_t stream) {
  const float* x  = (const float*)d_in[0];
  const float* W1 = (const float*)d_in[1];
  const float* b1 = (const float*)d_in[2];
  const float* W2 = (const float*)d_in[3];
  const float* b2 = (const float*)d_in[4];
  const float* W3 = (const float*)d_in[5];
  const float* b3 = (const float*)d_in[6];
  float* out = (float*)d_out;

  char* ws = (char*)d_ws;
  __bf16* qbuf  = (__bf16*)ws;               // 8000*128*2 = 2,048,000 B
  __bf16* vtbuf = (__bf16*)(ws + 2048000);   // 2,048,000 B
  float*  ctx   = (float*)(ws + 4096000);    // 4,096,000 B

  dim3 grid((N_POS + 127) / 128), block(256);
  proj_kernel<<<grid, block, 0, stream>>>(x, W1, b1, qbuf, 0);
  proj_kernel<<<grid, block, 0, stream>>>(x, W2, b2, vtbuf, 1);
  attn_kernel<<<grid, block, 0, stream>>>(qbuf, vtbuf, ctx);
  out_kernel<<<grid, block, 0, stream>>>(x, ctx, W3, b3, out);
}
